// AttnDecoderBlock_6579889897753
// MI455X (gfx1250) — matrix operand order repaired and checked by reading, not compiled, not run on hardware
//
#include <hip/hip_runtime.h>
#include <hip/hip_bf16.h>

// Problem sizes (fixed by the reference)
#define BB 64
#define NN 4096
#define MM 64
#define EE 256
#define CC 128

// Padded LDS row strides (in ushorts): keep 16B alignment for b128 frag loads
// while breaking the 64-bank power-of-two conflict pattern (2-way max).
#define SKS 264   // proj: s_k  [64][264]  (row = m, col = e)
#define WTS 264   // proj: s_wt [128][264] (row = c, col = e)
#define SKP 136   // attn: s_kp [64][136]  (row = m, col = c)
#define SQS 136   // attn: s_q  rows
#define SVP 72    // attn: s_vp [128][72]  (row = c, col = m)
#define SPS 72    // attn: s_p  rows

typedef __attribute__((ext_vector_type(16))) __bf16 v16bf;
typedef __attribute__((ext_vector_type(8)))  float  v8f;
typedef int async_v4i __attribute__((vector_size(16)));   // matches builtin param

#if __has_builtin(__builtin_amdgcn_global_load_async_to_lds_b128)
#define HAVE_ASYNC_LDS 1
#else
#define HAVE_ASYNC_LDS 0
#endif

union Frag {
    v16bf v;
    uint4 q[2];
};

__device__ __forceinline__ unsigned short f2bf(float f) {
    unsigned u = __float_as_uint(f);
    u += 0x7FFFu + ((u >> 16) & 1u);   // round-to-nearest-even
    return (unsigned short)(u >> 16);
}

__device__ __forceinline__ void wait_async_zero() {
#if __has_builtin(__builtin_amdgcn_s_wait_asynccnt)
    __builtin_amdgcn_s_wait_asynccnt(0);
#else
    asm volatile("s_wait_asynccnt 0" ::: "memory");
#endif
}

// ---------------------------------------------------------------------------
// Kernel 1: per-batch projections, WMMA version.
//   kp[b][m][c]  = (LN(key[b]) @ Wk)[m][c]      (bf16, row-major over C)
//   vpT[b][c][m] = (value[b]  @ Wv)[m][c]       (bf16, transposed: contiguous M)
// One block (256 threads = 8 waves) per batch.
// ---------------------------------------------------------------------------
__global__ void __launch_bounds__(256, 1)
proj_kernel(const float* __restrict__ key,
            const float* __restrict__ value,
            const float* __restrict__ wk,
            const float* __restrict__ wv,
            const float* __restrict__ g1,
            const float* __restrict__ b1,
            unsigned short* __restrict__ kp,
            unsigned short* __restrict__ vpT)
{
    __shared__ __align__(16) unsigned short s_k [MM * SKS];   // ~33 KB  A operand
    __shared__ __align__(16) unsigned short s_wt[CC * WTS];   // ~66 KB  B operand (W^T)

    const int b    = blockIdx.x;
    const int tid  = threadIdx.x;
    const int wave = tid >> 5;
    const int lane = tid & 31;
    const int half = lane >> 4;
    const int l16  = lane & 15;

    const int mrow0 = (wave & 3) * 16;   // WMMA: output row tile per wave
    const int col0  = (wave >> 2) * 64;  // WMMA: output col range per wave

    // ---- Phase A: LayerNorm(key[b]) over E -> s_k (bf16), 4 threads/row ----
    {
        const int row  = tid >> 2;
        const int part = tid & 3;
        const float* kr = key + ((size_t)b * MM + row) * EE + part * 64;
        float x[64];
        float s = 0.f;
        #pragma unroll
        for (int i = 0; i < 64; ++i) { x[i] = kr[i]; s += x[i]; }
        s += __shfl_xor(s, 1, 32);
        s += __shfl_xor(s, 2, 32);
        const float mu = s * (1.f / EE);
        float vs = 0.f;
        #pragma unroll
        for (int i = 0; i < 64; ++i) { float d = x[i] - mu; vs += d * d; }
        vs += __shfl_xor(vs, 1, 32);
        vs += __shfl_xor(vs, 2, 32);
        const float rstd = rsqrtf(vs * (1.f / EE) + 1e-5f);
        #pragma unroll
        for (int i = 0; i < 64; i += 2) {
            const int e = part * 64 + i;
            const unsigned lo = f2bf((x[i]     - mu) * rstd * g1[e]     + b1[e]);
            const unsigned hi = f2bf((x[i + 1] - mu) * rstd * g1[e + 1] + b1[e + 1]);
            *(unsigned*)&s_k[row * SKS + e] = lo | (hi << 16);
        }
    }
    // ---- Phase B: Wk^T -> s_wt (bf16) ----
    {
        const int c_ = tid & 127;
        const int eb = (tid >> 7) * 4;
        for (int e0 = eb; e0 < EE; e0 += 8) {
            const unsigned u0 = f2bf(wk[(e0 + 0) * CC + c_]);
            const unsigned u1 = f2bf(wk[(e0 + 1) * CC + c_]);
            const unsigned u2 = f2bf(wk[(e0 + 2) * CC + c_]);
            const unsigned u3 = f2bf(wk[(e0 + 3) * CC + c_]);
            uint2 pk;
            pk.x = u0 | (u1 << 16);
            pk.y = u2 | (u3 << 16);
            *(uint2*)&s_wt[c_ * WTS + e0] = pk;
        }
    }
    __syncthreads();

    // ---- Phase C: kp = LN(key) @ Wk  via WMMA (16x64 tile per wave) ----
    {
        v8f acc[4];
        #pragma unroll
        for (int i = 0; i < 4; ++i) acc[i] = (v8f){};
        #pragma unroll
        for (int kc = 0; kc < 8; ++kc) {
            const int e0 = kc * 32;
            Frag A;
            const unsigned short* ab = &s_k[(mrow0 + l16) * SKS + e0 + 16 * half];
            A.q[0] = *(const uint4*)ab;
            A.q[1] = *(const uint4*)(ab + 8);
            #pragma unroll
            for (int ct = 0; ct < 4; ++ct) {
                Frag Bf;
                const unsigned short* bb = &s_wt[(col0 + ct * 16 + l16) * WTS + e0 + 16 * half];
                Bf.q[0] = *(const uint4*)bb;
                Bf.q[1] = *(const uint4*)(bb + 8);
                acc[ct] = __builtin_amdgcn_wmma_f32_16x16x32_bf16(
                    false, A.v, false, Bf.v, (short)0, acc[ct], false, false);
            }
        }
        #pragma unroll
        for (int ct = 0; ct < 4; ++ct)
            #pragma unroll
            for (int r = 0; r < 8; ++r)
                kp[((size_t)b * MM + mrow0 + r + 8 * half) * CC + col0 + ct * 16 + l16] =
                    f2bf(acc[ct][r]);
    }
    __syncthreads();

    // ---- Phase D: value[b] -> s_k (bf16) ----
    {
        const float4* src = (const float4*)(value + (size_t)b * MM * EE);
        for (int i = tid; i < MM * EE / 4; i += 256) {
            const float4 xv = src[i];
            const int row = (i * 4) >> 8;      // / EE
            const int col = (i * 4) & 255;     // % EE
            uint2 pk;
            pk.x = f2bf(xv.x) | ((unsigned)f2bf(xv.y) << 16);
            pk.y = f2bf(xv.z) | ((unsigned)f2bf(xv.w) << 16);
            *(uint2*)&s_k[row * SKS + col] = pk;
        }
    }
    // ---- Phase E: Wv^T -> s_wt (bf16) ----
    {
        const int c_ = tid & 127;
        const int eb = (tid >> 7) * 4;
        for (int e0 = eb; e0 < EE; e0 += 8) {
            const unsigned u0 = f2bf(wv[(e0 + 0) * CC + c_]);
            const unsigned u1 = f2bf(wv[(e0 + 1) * CC + c_]);
            const unsigned u2 = f2bf(wv[(e0 + 2) * CC + c_]);
            const unsigned u3 = f2bf(wv[(e0 + 3) * CC + c_]);
            uint2 pk;
            pk.x = u0 | (u1 << 16);
            pk.y = u2 | (u3 << 16);
            *(uint2*)&s_wt[c_ * WTS + e0] = pk;
        }
    }
    __syncthreads();

    // ---- Phase F: vp = value @ Wv via WMMA; store transposed ----
    {
        v8f acc[4];
        #pragma unroll
        for (int i = 0; i < 4; ++i) acc[i] = (v8f){};
        #pragma unroll
        for (int kc = 0; kc < 8; ++kc) {
            const int e0 = kc * 32;
            Frag A;
            const unsigned short* ab = &s_k[(mrow0 + l16) * SKS + e0 + 16 * half];
            A.q[0] = *(const uint4*)ab;
            A.q[1] = *(const uint4*)(ab + 8);
            #pragma unroll
            for (int ct = 0; ct < 4; ++ct) {
                Frag Bf;
                const unsigned short* bb = &s_wt[(col0 + ct * 16 + l16) * WTS + e0 + 16 * half];
                Bf.q[0] = *(const uint4*)bb;
                Bf.q[1] = *(const uint4*)(bb + 8);
                acc[ct] = __builtin_amdgcn_wmma_f32_16x16x32_bf16(
                    false, A.v, false, Bf.v, (short)0, acc[ct], false, false);
            }
        }
        #pragma unroll
        for (int ct = 0; ct < 4; ++ct)
            #pragma unroll
            for (int r = 0; r < 8; ++r)
                vpT[((size_t)b * CC + col0 + ct * 16 + l16) * MM + mrow0 + r + 8 * half] =
                    f2bf(acc[ct][r]);
    }
}

// ---------------------------------------------------------------------------
// Kernel 2: fused LN(query) -> scores -> softmax -> output.
// 256 threads = 8 waves; each wave owns a 16-row query tile (block: 128 rows).
// grid = B * (N/128) = 2048 blocks. K/V tiles staged via async global->LDS
// (overlapped with the q-tile LayerNorm) into bank-conflict-padded buffers.
// ---------------------------------------------------------------------------
__global__ void __launch_bounds__(256)
attn_kernel(const float* __restrict__ query,
            const float* __restrict__ g2,
            const float* __restrict__ b2,
            const unsigned short* __restrict__ kp,
            const unsigned short* __restrict__ vpT,
            float* __restrict__ out)
{
    __shared__ __align__(16) unsigned short s_kp[MM * SKP];        // ~17 KB  [m][c]
    __shared__ __align__(16) unsigned short s_vp[CC * SVP];        // ~18 KB  [c][m]
    __shared__ __align__(16) unsigned short s_q [8][16 * SQS];     // ~34 KB
    __shared__ __align__(16) unsigned short s_p [8][16 * SPS];     // ~18 KB

    const int bt   = blockIdx.x;
    const int b    = bt >> 5;          // / (N/128)
    const int tile = bt & 31;
    const int row0 = tile * 128;
    const int tid  = threadIdx.x;
    const int wave = tid >> 5;         // 0..7
    const int lane = tid & 31;
    const int half = lane >> 4;        // which 16-lane half
    const int l16  = lane & 15;

    // ---- stage kp / vpT for this batch into LDS (async, overlapped w/ LN) ----
    {
        const uint4* gk = (const uint4*)(kp  + (size_t)b * MM * CC);
        const uint4* gv = (const uint4*)(vpT + (size_t)b * CC * MM);
        for (int i = tid; i < MM * CC / 8; i += 256) {
            // kp rows are 128 ushorts = 16 uint4; vp rows are 64 ushorts = 8 uint4
            uint4* dk = (uint4*)&s_kp[(i >> 4) * SKP + (i & 15) * 8];
            uint4* dv = (uint4*)&s_vp[(i >> 3) * SVP + (i & 7) * 8];
#if HAVE_ASYNC_LDS
            __builtin_amdgcn_global_load_async_to_lds_b128(
                (async_v4i*)(gk + i), (async_v4i*)dk, 0, 0);
            __builtin_amdgcn_global_load_async_to_lds_b128(
                (async_v4i*)(gv + i), (async_v4i*)dv, 0, 0);
#else
            *dk = gk[i];
            *dv = gv[i];
#endif
        }
    }

    // ---- LayerNorm the 16x128 q tile (one row at a time, full-wave reduce) ----
    for (int r = 0; r < 16; ++r) {
        const int grow = row0 + wave * 16 + r;
        const float* qr = query + ((size_t)b * NN + grow) * CC + lane * 4;
        float x0 = qr[0], x1 = qr[1], x2 = qr[2], x3 = qr[3];
        float s = x0 + x1 + x2 + x3;
        #pragma unroll
        for (int msk = 16; msk >= 1; msk >>= 1) s += __shfl_xor(s, msk, 32);
        const float mu = s * (1.f / CC);
        const float d0 = x0 - mu, d1 = x1 - mu, d2 = x2 - mu, d3 = x3 - mu;
        float vs = d0 * d0 + d1 * d1 + d2 * d2 + d3 * d3;
        #pragma unroll
        for (int msk = 16; msk >= 1; msk >>= 1) vs += __shfl_xor(vs, msk, 32);
        const float rstd = rsqrtf(vs * (1.f / CC) + 1e-5f);
        const int cb = lane * 4;
        const unsigned y0 = f2bf(d0 * rstd * g2[cb + 0] + b2[cb + 0]);
        const unsigned y1 = f2bf(d1 * rstd * g2[cb + 1] + b2[cb + 1]);
        const unsigned y2 = f2bf(d2 * rstd * g2[cb + 2] + b2[cb + 2]);
        const unsigned y3 = f2bf(d3 * rstd * g2[cb + 3] + b2[cb + 3]);
        uint2 pk;
        pk.x = y0 | (y1 << 16);
        pk.y = y2 | (y3 << 16);
        *(uint2*)&s_q[wave][r * SQS + cb] = pk;
    }
#if HAVE_ASYNC_LDS
    wait_async_zero();     // our async-to-LDS transfers have landed
#endif
    __syncthreads();       // everyone's transfers + LN stores visible

    // ---- S[16 x 64] = q_ln @ kp^T  (K = C = 128: 4 chunks of 32) ----
    v8f S[4];
    #pragma unroll
    for (int i = 0; i < 4; ++i) S[i] = (v8f){};
    #pragma unroll
    for (int kc = 0; kc < 4; ++kc) {
        const int c0 = kc * 32;
        Frag A;  // A layout: row = l16, 16 contiguous K per lane (lo/hi halves)
        const unsigned short* qb = &s_q[wave][l16 * SQS + c0 + 16 * half];
        A.q[0] = *(const uint4*)qb;
        A.q[1] = *(const uint4*)(qb + 8);
        #pragma unroll
        for (int mt = 0; mt < 4; ++mt) {
            Frag Bf; // B layout: col = l16, 16 contiguous K per lane (lo/hi halves)
            const unsigned short* kb = &s_kp[(mt * 16 + l16) * SKP + c0 + 16 * half];
            Bf.q[0] = *(const uint4*)kb;
            Bf.q[1] = *(const uint4*)(kb + 8);
            S[mt] = __builtin_amdgcn_wmma_f32_16x16x32_bf16(
                false, A.v, false, Bf.v, (short)0, S[mt], false, false);
        }
    }

    // ---- softmax over M=64 (row = reg + 8*half; 16-lane + 4-tile reduce) ----
    #pragma unroll
    for (int r = 0; r < 8; ++r) {
        float mx = fmaxf(fmaxf(S[0][r], S[1][r]), fmaxf(S[2][r], S[3][r]));
        #pragma unroll
        for (int msk = 8; msk >= 1; msk >>= 1) mx = fmaxf(mx, __shfl_xor(mx, msk, 32));
        const float e0 = __expf(S[0][r] - mx);
        const float e1 = __expf(S[1][r] - mx);
        const float e2 = __expf(S[2][r] - mx);
        const float e3 = __expf(S[3][r] - mx);
        float sum = e0 + e1 + e2 + e3;
        #pragma unroll
        for (int msk = 8; msk >= 1; msk >>= 1) sum += __shfl_xor(sum, msk, 32);
        const float inv = 1.f / sum;
        const int prow = r + 8 * half;
        s_p[wave][prow * SPS +  0 + l16] = f2bf(e0 * inv);
        s_p[wave][prow * SPS + 16 + l16] = f2bf(e1 * inv);
        s_p[wave][prow * SPS + 32 + l16] = f2bf(e2 * inv);
        s_p[wave][prow * SPS + 48 + l16] = f2bf(e3 * inv);
    }
    __syncthreads();

    // ---- O[16 x 128] = P @ V'  (K = M = 64: 2 chunks of 32; 8 col tiles) ----
    v8f O[8];
    #pragma unroll
    for (int i = 0; i < 8; ++i) O[i] = (v8f){};
    #pragma unroll
    for (int kc = 0; kc < 2; ++kc) {
        const int k0 = kc * 32;
        Frag A;
        const unsigned short* pb = &s_p[wave][l16 * SPS + k0 + 16 * half];
        A.q[0] = *(const uint4*)pb;
        A.q[1] = *(const uint4*)(pb + 8);
        #pragma unroll
        for (int ct = 0; ct < 8; ++ct) {
            Frag Bf;
            const unsigned short* vb = &s_vp[(ct * 16 + l16) * SVP + k0 + 16 * half];
            Bf.q[0] = *(const uint4*)vb;
            Bf.q[1] = *(const uint4*)(vb + 8);
            O[ct] = __builtin_amdgcn_wmma_f32_16x16x32_bf16(
                false, A.v, false, Bf.v, (short)0, O[ct], false, false);
        }
    }

    // ---- store O (f32) ----
    #pragma unroll
    for (int ct = 0; ct < 8; ++ct) {
        #pragma unroll
        for (int r = 0; r < 8; ++r) {
            const int orow = r + 8 * half;
            const int ocol = ct * 16 + l16;
            out[((size_t)b * NN + row0 + wave * 16 + orow) * CC + ocol] = O[ct][r];
        }
    }
}

// ---------------------------------------------------------------------------
extern "C" void kernel_launch(void* const* d_in, const int* in_sizes, int n_in,
                              void* d_out, int out_size, void* d_ws, size_t ws_size,
                              hipStream_t stream) {
    const float* query = (const float*)d_in[0];
    const float* key   = (const float*)d_in[1];
    const float* value = (const float*)d_in[2];
    const float* wk    = (const float*)d_in[3];
    const float* wv    = (const float*)d_in[4];
    const float* g1    = (const float*)d_in[5];
    const float* b1    = (const float*)d_in[6];
    const float* g2    = (const float*)d_in[7];
    const float* b2    = (const float*)d_in[8];
    float* out = (float*)d_out;

    unsigned short* kp  = (unsigned short*)d_ws;                  // B*M*C bf16 (1 MB)
    unsigned short* vpT = kp + (size_t)BB * MM * CC;              // B*C*M bf16 (1 MB)

    proj_kernel<<<BB, 256, 0, stream>>>(key, value, wk, wv, g1, b1, kp, vpT);
    attn_kernel<<<BB * (NN / 128), 256, 0, stream>>>(query, g2, b2, kp, vpT, out);
}
